// Attention_75892072120962
// MI455X (gfx1250) — matrix operand order repaired and checked by reading, not compiled, not run on hardware
//
#include <hip/hip_runtime.h>
#include <hip/hip_bf16.h>

// ---------------------------------------------------------------------------
// Problem constants (match reference)
// ---------------------------------------------------------------------------
constexpr int T_  = 2048;
constexpr int D_  = 4096;
constexpr int NQ  = 32;          // query heads
constexpr int KH  = 8;           // kv heads
constexpr int H_  = 128;
constexpr int NH_ = NQ * H_;     // 4096
constexpr int KVD = KH * H_;     // 1024
constexpr float SM_SCALE = 0.08838834764831845f;  // 128^-0.5
constexpr float NEG_INF  = -1e30f;

typedef __attribute__((ext_vector_type(16))) __bf16 v16bf;
typedef __attribute__((ext_vector_type(8)))  float  v8f;

union Frag16 { v16bf v; uint4 u[2]; };

__device__ __forceinline__ unsigned short f32_to_bf16(float f) {
  unsigned int x = __builtin_bit_cast(unsigned int, f);
  unsigned int r = x + 0x7FFFu + ((x >> 16) & 1u);   // round-to-nearest-even
  return (unsigned short)(r >> 16);
}

// Async global->LDS 16-byte copy (CDNA5, tracked by ASYNCcnt).
// vdst = per-lane LDS byte address, vaddr = per-lane u32 byte offset, saddr = base.
__device__ __forceinline__ void async_copy_b128(unsigned lds_addr,
                                                const void* base,
                                                unsigned byte_off) {
  asm volatile("global_load_async_to_lds_b128 %0, %1, %2"
               :
               : "v"(lds_addr), "v"(byte_off), "s"(base)
               : "memory");
}
__device__ __forceinline__ void wait_async_zero() {
  asm volatile("s_wait_asynccnt 0x0" ::: "memory");
}

// Low 32 bits of a generic pointer to __shared__ = LDS byte address
__device__ __forceinline__ unsigned lds_addr_of(const void* p) {
  return (unsigned)(unsigned long long)p;
}

// ---------------------------------------------------------------------------
// f32 -> bf16 conversion (row-major copy)
// ---------------------------------------------------------------------------
__global__ __launch_bounds__(256) void cvt_bf16_kernel(const float* __restrict__ src,
                                                       unsigned short* __restrict__ dst,
                                                       int n) {
  int i = blockIdx.x * 256 + threadIdx.x;
  if (i < n) dst[i] = f32_to_bf16(src[i]);
}

// ---------------------------------------------------------------------------
// f32 [R,C] -> bf16 transposed [C,R], tiled through LDS (coalesced both ways)
// grid: (C/64, R/64), 256 threads
// ---------------------------------------------------------------------------
__global__ __launch_bounds__(256) void cvt_t_bf16_kernel(const float* __restrict__ src,
                                                         unsigned short* __restrict__ dst,
                                                         int R, int Cc) {
  __shared__ unsigned short tile[64][65];
  int r0 = blockIdx.y * 64, c0 = blockIdx.x * 64;
#pragma unroll
  for (int i = 0; i < 16; ++i) {
    int e = threadIdx.x + i * 256;          // 0..4095
    int rr = e >> 6, cc = e & 63;
    tile[rr][cc] = f32_to_bf16(src[(size_t)(r0 + rr) * Cc + c0 + cc]);
  }
  __syncthreads();
#pragma unroll
  for (int i = 0; i < 16; ++i) {
    int e = threadIdx.x + i * 256;
    int cc = e >> 6, rr = e & 63;
    dst[(size_t)(c0 + cc) * R + r0 + rr] = tile[rr][cc];
  }
}

// ---------------------------------------------------------------------------
// GEMM: C[M,Nd] (f32) = A[M,Kd] (bf16 row-major) * Bt[Nd,Kd] (bf16 row-major,
// i.e. B pre-transposed). Compile-time Kd/Nd so strides fold into offsets.
// Block: 256 threads = 8 waves (2x4), tile 64(M) x 128(N), K-chunk 64.
// A/B tiles staged with GLOBAL_LOAD_ASYNC_TO_LDS_B128.
// ---------------------------------------------------------------------------
#define GT_M 64
#define GT_N 128
#define GT_K 64

template <int Kd, int Nd>
__global__ __launch_bounds__(256) void gemm_bf16_kernel(
    const unsigned short* __restrict__ A,
    const unsigned short* __restrict__ Bt,
    float* __restrict__ C) {
  __shared__ unsigned short a_lds[GT_M * GT_K];   // 8 KB
  __shared__ unsigned short b_lds[GT_N * GT_K];   // 16 KB, [col][k]
  const int tid  = threadIdx.x;
  const int lane = tid & 31;
  const int w    = tid >> 5;
  const int m0   = blockIdx.y * GT_M;
  const int n0   = blockIdx.x * GT_N;
  const int wm   = w >> 2;       // 0..1
  const int wn   = w & 3;        // 0..3
  const int r16  = lane & 15;
  const int half = lane >> 4;

  const unsigned a_base = lds_addr_of(a_lds);
  const unsigned b_base = lds_addr_of(b_lds);

  v8f acc[2][2] = {};

  for (int k0 = 0; k0 < Kd; k0 += GT_K) {
    // --- async stage A tile [64 x 64] : 512 segs of 16B, 2 per thread ---
#pragma unroll
    for (int i = 0; i < 2; ++i) {
      int s = tid + i * 256;
      int row = s >> 3, off = (s & 7) * 8;
      async_copy_b128(a_base + (unsigned)(row * GT_K + off) * 2, A,
                      (unsigned)((m0 + row) * Kd + k0 + off) * 2);
    }
    // --- async stage Bt tile [128 x 64] : 1024 segs of 16B, 4 per thread ---
#pragma unroll
    for (int i = 0; i < 4; ++i) {
      int s = tid + i * 256;
      int col = s >> 3, off = (s & 7) * 8;
      async_copy_b128(b_base + (unsigned)(col * GT_K + off) * 2, Bt,
                      (unsigned)((n0 + col) * Kd + k0 + off) * 2);
    }
    // prefetch next chunk into L2
    __builtin_prefetch(A + (size_t)(m0 + (tid >> 3)) * Kd + k0 + GT_K, 0, 1);
    __builtin_prefetch(Bt + (size_t)(n0 + (tid >> 1)) * Kd + k0 + GT_K, 0, 1);

    wait_async_zero();
    __syncthreads();

    // --- two K-steps of 32 ---
#pragma unroll
    for (int ks = 0; ks < GT_K; ks += 32) {
      Frag16 af[2], bf2[2];
#pragma unroll
      for (int mt = 0; mt < 2; ++mt) {
        const unsigned short* p = a_lds + (wm * 32 + mt * 16 + r16) * GT_K + ks;
        af[mt].u[0] = *(const uint4*)(p + half * 8);        // K = half*8+0..7
        af[mt].u[1] = *(const uint4*)(p + 16 + half * 8);   // K = 16+half*8+0..7
      }
#pragma unroll
      for (int nt = 0; nt < 2; ++nt) {
        const unsigned short* p = b_lds + (wn * 32 + nt * 16 + r16) * GT_K + ks + half * 8;
        bf2[nt].u[0] = *(const uint4*)(p);                  // K = half*8+0..7
        bf2[nt].u[1] = *(const uint4*)(p + 16);             // K = 16+half*8+0..7
      }
#pragma unroll
      for (int mt = 0; mt < 2; ++mt)
#pragma unroll
        for (int nt = 0; nt < 2; ++nt)
          acc[mt][nt] = __builtin_amdgcn_wmma_f32_16x16x32_bf16(
              false, af[mt].v, false, bf2[nt].v, (short)0, acc[mt][nt], false, false);
    }
    __syncthreads();
  }

  // --- store C: vgpr j holds row (j + 8*half), col r16; Nd constant folds ---
#pragma unroll
  for (int mt = 0; mt < 2; ++mt)
#pragma unroll
    for (int nt = 0; nt < 2; ++nt) {
      float* cb = C + (size_t)(m0 + wm * 32 + mt * 16 + 8 * half) * Nd
                    + n0 + wn * 32 + nt * 16 + r16;
#pragma unroll
      for (int j = 0; j < 8; ++j)
        cb[j * Nd] = acc[mt][nt][j];
    }
}

// ---------------------------------------------------------------------------
// RoPE on Q (f32 in, bf16 out)
// ---------------------------------------------------------------------------
__global__ __launch_bounds__(256) void rope_q_kernel(const float* __restrict__ qf,
                                                     const int* __restrict__ positions,
                                                     unsigned short* __restrict__ qb) {
  int idx = blockIdx.x * 256 + threadIdx.x;           // T * N * 64
  if (idx >= T_ * NQ * 64) return;
  int h = idx & 63, tn = idx >> 6;
  int n = tn & (NQ - 1), t = tn / NQ;
  float inv_freq = __powf(10000.0f, -(float)(2 * h) / (float)H_);
  float ang = (float)positions[t] * inv_freq;
  float s, c;
  __sincosf(ang, &s, &c);
  const float* base = qf + (size_t)t * NH_ + n * H_;
  float x1 = base[h], x2 = base[h + 64];
  unsigned short* ob = qb + (size_t)t * NH_ + n * H_;
  ob[h]      = f32_to_bf16(x1 * c - x2 * s);
  ob[h + 64] = f32_to_bf16(x2 * c + x1 * s);
}

// ---------------------------------------------------------------------------
// RoPE on K + int8 quantize; write f32 cache (d_out) and bf16 copy [T,KH,H]
// ---------------------------------------------------------------------------
__global__ __launch_bounds__(256) void rope_k_quant_kernel(const float* __restrict__ kf,
                                                           const int* __restrict__ positions,
                                                           float* __restrict__ kcache,
                                                           unsigned short* __restrict__ kb) {
  int idx = blockIdx.x * 256 + threadIdx.x;           // T * KH * 64
  if (idx >= T_ * KH * 64) return;
  int h = idx & 63, tk = idx >> 6;
  int k = tk & (KH - 1), t = tk / KH;
  int pos = positions[t];
  float inv_freq = __powf(10000.0f, -(float)(2 * h) / (float)H_);
  float ang = (float)pos * inv_freq;
  float s, c;
  __sincosf(ang, &s, &c);
  const float* base = kf + (size_t)t * KVD + k * H_;
  float x1 = base[h], x2 = base[h + 64];
  float o1 = x1 * c - x2 * s;
  float o2 = x2 * c + x1 * s;
  float q1 = fminf(fmaxf(rintf(o1), -128.0f), 127.0f);   // K_SCALE == 1
  float q2 = fminf(fmaxf(rintf(o2), -128.0f), 127.0f);
  size_t row = (size_t)pos * KVD + k * H_;
  kcache[row + h]      = q1;
  kcache[row + h + 64] = q2;
  kb[row + h]      = f32_to_bf16(q1);                    // exact (|v| <= 128)
  kb[row + h + 64] = f32_to_bf16(q2);
}

// ---------------------------------------------------------------------------
// V quantize; write f32 cache (d_out) and bf16 transposed copy [KH,H,T]
// ---------------------------------------------------------------------------
__global__ __launch_bounds__(256) void quant_v_kernel(const float* __restrict__ vf,
                                                      const int* __restrict__ positions,
                                                      float* __restrict__ vcache,
                                                      unsigned short* __restrict__ vt) {
  int idx = blockIdx.x * 256 + threadIdx.x;           // T * KVD
  if (idx >= T_ * KVD) return;
  int h = idx & (H_ - 1);
  int tk = idx >> 7;
  int k = tk & (KH - 1), t = tk / KH;
  int pos = positions[t];
  float q = fminf(fmaxf(rintf(vf[(size_t)t * KVD + k * H_ + h]), -128.0f), 127.0f);
  vcache[(size_t)pos * KVD + k * H_ + h] = q;
  vt[((size_t)k * H_ + h) * T_ + pos] = f32_to_bf16(q);  // exact
}

// ---------------------------------------------------------------------------
// Flash attention (causal, GQA). One wave per (16-row Q tile, head).
// q_bf [T,NH]; k_bf [T,KH,H] (K^T frags contiguous); vt_bf [KH,H,T].
// ---------------------------------------------------------------------------
__global__ __launch_bounds__(256) void attn_kernel(const unsigned short* __restrict__ qb,
                                                   const unsigned short* __restrict__ kb,
                                                   const unsigned short* __restrict__ vtb,
                                                   const int* __restrict__ positions,
                                                   unsigned short* __restrict__ ob) {
  __shared__ unsigned short p_lds[8][16 * 32];        // per-wave P patch
  const int lane = threadIdx.x & 31;
  const int w    = threadIdx.x >> 5;
  const int widx = blockIdx.x * 8 + w;
  const int tile = widx >> 5;                         // widx / NQ
  const int head = widx & (NQ - 1);
  const int t0   = tile * 16;
  const int kvh  = head >> 2;                         // head / (NQ/KH)
  const int r16  = lane & 15;
  const int half = lane >> 4;

  Frag16 qf[4];
  const unsigned short* qrow = qb + (size_t)(t0 + r16) * NH_ + head * H_;
#pragma unroll
  for (int kc = 0; kc < 4; ++kc) {
    qf[kc].u[0] = *(const uint4*)(qrow + kc * 32 + half * 8);
    qf[kc].u[1] = *(const uint4*)(qrow + kc * 32 + 16 + half * 8);
  }

  float mrow[8], lrow[8];
  int   prow[8];
  int   pmax = 0;
  v8f   oacc[8] = {};
#pragma unroll
  for (int j = 0; j < 8; ++j) {
    mrow[j] = NEG_INF;
    lrow[j] = 0.0f;
    prow[j] = positions[t0 + j + 8 * half];
    pmax = max(pmax, prow[j]);
  }
#pragma unroll
  for (int off = 16; off >= 1; off >>= 1)
    pmax = max(pmax, __shfl_xor(pmax, off, 32));

  unsigned short* pl = p_lds[w];

  for (int s0 = 0; s0 <= pmax; s0 += 32) {
    // --- scores: S[16,32] = Q (16x128) * K^T (128x32) ---
    v8f sc[2] = {};
#pragma unroll
    for (int kc = 0; kc < 4; ++kc) {
#pragma unroll
      for (int st = 0; st < 2; ++st) {
        Frag16 kfrag;
        const int s = s0 + st * 16 + r16;             // B-frag column = key index
        const unsigned short* kp = kb + (size_t)s * KVD + kvh * H_ + kc * 32 + half * 8;
        kfrag.u[0] = *(const uint4*)(kp);
        kfrag.u[1] = *(const uint4*)(kp + 16);
        __builtin_prefetch(kp + 32 * KVD, 0, 1);      // next key block
        sc[st] = __builtin_amdgcn_wmma_f32_16x16x32_bf16(
            false, qf[kc].v, false, kfrag.v, (short)0, sc[st], false, false);
      }
    }

    // --- online softmax on C-layout (row = j+8*half, col = r16) ---
#pragma unroll
    for (int j = 0; j < 8; ++j) {
      float v0 = sc[0][j] * SM_SCALE;
      float v1 = sc[1][j] * SM_SCALE;
      if (s0 + r16 > prow[j])      v0 = NEG_INF;      // causal mask
      if (s0 + 16 + r16 > prow[j]) v1 = NEG_INF;
      float m = fmaxf(v0, v1);
#pragma unroll
      for (int off = 8; off >= 1; off >>= 1)
        m = fmaxf(m, __shfl_xor(m, off, 32));
      float mc    = fmaxf(mrow[j], m);
      float alpha = __expf(mrow[j] - mc);
      mrow[j] = mc;
      float p0 = __expf(v0 - mc);
      float p1 = __expf(v1 - mc);
      float ls = p0 + p1;
#pragma unroll
      for (int off = 8; off >= 1; off >>= 1)
        ls += __shfl_xor(ls, off, 32);
      lrow[j] = lrow[j] * alpha + ls;
#pragma unroll
      for (int hc = 0; hc < 8; ++hc) oacc[hc][j] *= alpha;
      int row = j + 8 * half;
      pl[row * 32 + r16]      = f32_to_bf16(p0);
      pl[row * 32 + 16 + r16] = f32_to_bf16(p1);
    }

    __builtin_amdgcn_wave_barrier();
    asm volatile("s_wait_dscnt 0x0" ::: "memory");    // wave-local LDS visibility

    Frag16 pf;
    pf.u[0] = *(const uint4*)(pl + r16 * 32 + half * 8);
    pf.u[1] = *(const uint4*)(pl + r16 * 32 + 16 + half * 8);
    __builtin_amdgcn_wave_barrier();

    // --- O += P (16x32) * V (32x128) ---
#pragma unroll
    for (int hc = 0; hc < 8; ++hc) {
      Frag16 vfrag;
      const unsigned short* vp =
          vtb + ((size_t)kvh * H_ + hc * 16 + r16) * T_ + s0 + half * 8;
      vfrag.u[0] = *(const uint4*)(vp);
      vfrag.u[1] = *(const uint4*)(vp + 16);
      __builtin_prefetch(vp + 32, 0, 1);              // next key block
      oacc[hc] = __builtin_amdgcn_wmma_f32_16x16x32_bf16(
          false, pf.v, false, vfrag.v, (short)0, oacc[hc], false, false);
    }
  }

  // --- normalize and store o (bf16) ---
#pragma unroll
  for (int j = 0; j < 8; ++j) {
    float inv = 1.0f / lrow[j];
    unsigned short* orow = ob + (size_t)(t0 + j + 8 * half) * NH_ + head * H_;
#pragma unroll
    for (int hc = 0; hc < 8; ++hc)
      orow[hc * 16 + r16] = f32_to_bf16(oacc[hc][j] * inv);
  }
}

// ---------------------------------------------------------------------------
// Host-side launch
// ---------------------------------------------------------------------------
extern "C" void kernel_launch(void* const* d_in, const int* in_sizes, int n_in,
                              void* d_out, int out_size, void* d_ws, size_t ws_size,
                              hipStream_t stream) {
  const float* x  = (const float*)d_in[0];
  const float* Wq = (const float*)d_in[1];
  const float* Wk = (const float*)d_in[2];
  const float* Wv = (const float*)d_in[3];
  const float* Wo = (const float*)d_in[4];
  const int* positions = (const int*)d_in[7];

  float* kc_out = (float*)d_out;                       // [T,KH,H] f32
  float* vc_out = kc_out + (size_t)T_ * KVD;           // [T,KH,H] f32
  float* o_out  = vc_out + (size_t)T_ * KVD;           // [T,D]    f32

  char* ws = (char*)d_ws;
  size_t off = 0;
  auto walloc = [&](size_t bytes) -> char* {
    char* p = ws + off;
    off += (bytes + 255) & ~(size_t)255;
    return p;
  };
  unsigned short* x_bf   = (unsigned short*)walloc((size_t)T_ * D_ * 2);
  unsigned short* wqt_bf = (unsigned short*)walloc((size_t)NH_ * D_ * 2);  // [NH,D]
  unsigned short* wkt_bf = (unsigned short*)walloc((size_t)KVD * D_ * 2);  // [KVD,D]
  unsigned short* wvt_bf = (unsigned short*)walloc((size_t)KVD * D_ * 2);  // [KVD,D]
  unsigned short* wot_bf = (unsigned short*)walloc((size_t)D_ * NH_ * 2);  // [D,NH]
  float*          q_f32  = (float*)walloc((size_t)T_ * NH_ * 4);
  float*          k_f32  = (float*)walloc((size_t)T_ * KVD * 4);
  float*          v_f32  = (float*)walloc((size_t)T_ * KVD * 4);
  unsigned short* q_bf   = (unsigned short*)walloc((size_t)T_ * NH_ * 2);
  unsigned short* k_bf   = (unsigned short*)walloc((size_t)T_ * KVD * 2);
  unsigned short* vt_bf  = (unsigned short*)walloc((size_t)KVD * T_ * 2);
  unsigned short* o_bf   = (unsigned short*)walloc((size_t)T_ * NH_ * 2);
  (void)ws_size; (void)in_sizes; (void)n_in; (void)out_size;

  // x: plain convert; weights: convert + transpose (B operands become [N,K])
  cvt_bf16_kernel<<<(T_ * D_ + 255) / 256, 256, 0, stream>>>(x, x_bf, T_ * D_);
  cvt_t_bf16_kernel<<<dim3(NH_ / 64, D_ / 64), 256, 0, stream>>>(Wq, wqt_bf, D_, NH_);
  cvt_t_bf16_kernel<<<dim3(KVD / 64, D_ / 64), 256, 0, stream>>>(Wk, wkt_bf, D_, KVD);
  cvt_t_bf16_kernel<<<dim3(KVD / 64, D_ / 64), 256, 0, stream>>>(Wv, wvt_bf, D_, KVD);
  cvt_t_bf16_kernel<<<dim3(D_ / 64, NH_ / 64), 256, 0, stream>>>(Wo, wot_bf, NH_, D_);

  // Projections: q = x*Wq, k = x*Wk, v = x*Wv   (Kd = D = 4096 for all)
  gemm_bf16_kernel<D_, NH_><<<dim3(NH_ / GT_N, T_ / GT_M), 256, 0, stream>>>(
      x_bf, wqt_bf, q_f32);
  gemm_bf16_kernel<D_, KVD><<<dim3(KVD / GT_N, T_ / GT_M), 256, 0, stream>>>(
      x_bf, wkt_bf, k_f32);
  gemm_bf16_kernel<D_, KVD><<<dim3(KVD / GT_N, T_ / GT_M), 256, 0, stream>>>(
      x_bf, wvt_bf, v_f32);

  // RoPE + quantize epilogues
  rope_q_kernel<<<(T_ * NQ * 64 + 255) / 256, 256, 0, stream>>>(q_f32, positions, q_bf);
  rope_k_quant_kernel<<<(T_ * KH * 64 + 255) / 256, 256, 0, stream>>>(
      k_f32, positions, kc_out, k_bf);
  quant_v_kernel<<<(T_ * KVD + 255) / 256, 256, 0, stream>>>(
      v_f32, positions, vc_out, vt_bf);

  // Flash attention: (T/16)*NQ waves, 8 waves per block
  attn_kernel<<<(T_ / 16) * NQ / 8, 256, 0, stream>>>(q_bf, k_bf, vt_bf, positions, o_bf);

  // Output projection: o_TD = o * Wo   (Kd = NH = 4096)
  gemm_bf16_kernel<NH_, D_><<<dim3(D_ / GT_N, T_ / GT_M), 256, 0, stream>>>(
      o_bf, wot_bf, o_out);
}
